// NearbyAttention_16441134809832
// MI455X (gfx1250) — matrix operand order repaired and checked by reading, not compiled, not run on hardware
//
#include <hip/hip_runtime.h>
#include <hip/hip_bf16.h>

// ---------------- problem constants ----------------
#define SEQ    2561          // real sequence length
#define SEQP   2688          // padded: 21 * 128 = 168 * 16 = 84 * 32
#define DIMK   1024          // model dim == inner dim
#define HEADS  16
#define DH     64
#define QT     168           // query tiles of 16
#define KB     84            // key blocks of 32

typedef __bf16 bf16_t;
typedef bf16_t v16bf __attribute__((ext_vector_type(16)));
typedef bf16_t v8bf  __attribute__((ext_vector_type(8)));
typedef float  v8f   __attribute__((ext_vector_type(8)));

static __device__ __forceinline__ v8f wmma_bf16(v16bf a, v16bf b, v8f c) {
    // D = A(16x32 bf16) * B(32x16 bf16) + C(16x16 f32)
    return __builtin_amdgcn_wmma_f32_16x16x32_bf16(
        /*neg_a=*/false, a, /*neg_b=*/false, b,
        /*c_mod=*/(short)0, c, /*reuse_a=*/false, /*reuse_b=*/false);
}

static __device__ __forceinline__ v16bf ld_frag(const bf16_t* p0, const bf16_t* p1) {
    v8bf a0 = *(const v8bf*)p0;
    v8bf a1 = *(const v8bf*)p1;
    v16bf a;
#pragma unroll
    for (int i = 0; i < 8; ++i) { a[i] = a0[i]; a[8 + i] = a1[i]; }
    return a;
}

// ---------------------------------------------------------------
// fp32 -> bf16 convert with row zero-padding
// ---------------------------------------------------------------
__global__ void k_pad_bf16(const float* __restrict__ in, bf16_t* __restrict__ out,
                           int rows_src, int rows_dst, int cols) {
    int idx = blockIdx.x * blockDim.x + threadIdx.x;
    int total = rows_dst * cols;
    if (idx >= total) return;
    int r = idx / cols;
    float v = (r < rows_src) ? in[idx] : 0.0f;
    out[idx] = (bf16_t)v;
}

// ---------------------------------------------------------------
// fp32 [k][n] -> bf16 transposed [n][k]  (1024 x 1024 weights)
// Puts weights in the exact WMMA B-operand order: lane = column n,
// K contiguous -> B fragments become two 16B global reads per lane.
// ---------------------------------------------------------------
__global__ void k_wtrans_bf16(const float* __restrict__ in, bf16_t* __restrict__ out) {
    int idx = blockIdx.x * blockDim.x + threadIdx.x;
    if (idx >= DIMK * DIMK) return;
    int n = idx >> 10;          // output row  (original column)
    int k = idx & (DIMK - 1);   // output col  (original row)
    out[idx] = (bf16_t)in[(size_t)k * DIMK + n];
}

// ---------------------------------------------------------------
// Block-level mask summary: bm[qt][kbi] = 1 if ANY (q,key) in the
// 16x32 tile is unmasked (mask byte == 0, q < SEQ, key < SEQ).
// ---------------------------------------------------------------
__global__ void k_maskblk(const unsigned char* __restrict__ mask,
                          unsigned char* __restrict__ bm) {
    int idx = blockIdx.x * blockDim.x + threadIdx.x;
    if (idx >= QT * KB) return;
    int qt  = idx / KB;
    int kbi = idx % KB;
    int q0  = qt * 16;
    int kb  = kbi * 32;
    bool any = false;
    for (int r = 0; r < 16 && !any; ++r) {
        int q = q0 + r;
        if (q >= SEQ) break;
        const unsigned char* row = mask + (size_t)q * SEQ + kb;
        int kmax = (kb + 32 <= SEQ) ? 32 : (SEQ - kb);
        for (int c = 0; c < kmax; ++c) {
            if (row[c] == 0) { any = true; break; }
        }
    }
    bm[idx] = any ? 1 : 0;
}

// ---------------------------------------------------------------
// LDS-free bf16 GEMM, f32 accumulate:
//   C[M x 1024] = A[M x 1024] * Bt^T   (Bt is B transposed, [n][k])
// block = 128 threads = 4 waves. Wave tile 32x64 (2 A-frags x 4 B-frags
// -> 8 WMMAs per k-step, 12 b128 loads). B panel reuse via WGP$/L2.
// mode 0: store bf16 head-split      out[(n>>6)*SEQP + m][DH]  (Q, K)
// mode 1: store bf16 head-transposed out[n*SEQP + m]           (V^T)
// mode 2: store f32 + bias, rows < SEQ, into d_out
// ---------------------------------------------------------------
__global__ __launch_bounds__(128)
void k_gemm(const bf16_t* __restrict__ A, const bf16_t* __restrict__ Bt,
            void* __restrict__ Out, const float* __restrict__ bias, int mode) {
    const int tid  = threadIdx.x;
    const int wave = tid >> 5;
    const int lane = tid & 31;
    const int l15  = lane & 15;
    const int hi   = lane >> 4;

    const int m0 = blockIdx.x * 128 + wave * 32;   // 32 rows per wave
    const int n0 = blockIdx.y * 64;

    const bf16_t* aRow0 = A + (size_t)(m0 + l15) * DIMK + hi * 8;
    const bf16_t* aRow1 = A + (size_t)(m0 + 16 + l15) * DIMK + hi * 8;

    v8f acc0[4] = {};
    v8f acc1[4] = {};

    for (int k0 = 0; k0 < DIMK; k0 += 32) {
        // prefetch next k-step into near cache
        if (k0 + 32 < DIMK) {
            __builtin_prefetch(aRow0 + k0 + 32, 0, 3);
            __builtin_prefetch(aRow1 + k0 + 32, 0, 3);
            __builtin_prefetch(Bt + (size_t)(n0 + l15 * 4) * DIMK + k0 + 32, 0, 3);
        }

        // A fragments: lane<16 holds K {0..7,16..23}, lane>=16 {8..15,24..31}
        v16bf a0 = ld_frag(aRow0 + k0, aRow0 + k0 + 16);
        v16bf a1 = ld_frag(aRow1 + k0, aRow1 + k0 + 16);

        // B fragments: lane<16 holds K {0..7,16..23}, lane>=16 {8..15,24..31}
        v16bf b[4];
#pragma unroll
        for (int j = 0; j < 4; ++j) {
            const bf16_t* bp = Bt + (size_t)(n0 + j * 16 + l15) * DIMK + k0 + hi * 8;
            b[j] = ld_frag(bp, bp + 16);
        }

#pragma unroll
        for (int j = 0; j < 4; ++j)
            acc0[j] = wmma_bf16(a0, b[j], acc0[j]);
#pragma unroll
        for (int j = 0; j < 4; ++j)
            acc1[j] = wmma_bf16(a1, b[j], acc1[j]);
    }

    // ---- store: C frag element (M = r + 8*hi, N = n0 + 16j + l15) ----
#pragma unroll
    for (int g = 0; g < 2; ++g) {
        const v8f* acc = g ? acc1 : acc0;
        const int mg = m0 + g * 16;
#pragma unroll
        for (int j = 0; j < 4; ++j) {
            int n = n0 + j * 16 + l15;
#pragma unroll
            for (int r = 0; r < 8; ++r) {
                int m = mg + r + hi * 8;
                float v = acc[j][r];
                if (mode == 0) {
                    bf16_t* o = (bf16_t*)Out;
                    o[((size_t)(n >> 6) * SEQP + m) * DH + (n & 63)] = (bf16_t)v;
                } else if (mode == 1) {
                    bf16_t* o = (bf16_t*)Out;
                    o[(size_t)n * SEQP + m] = (bf16_t)v;
                } else {
                    if (m < SEQ) {
                        float* o = (float*)Out;
                        o[(size_t)m * DIMK + n] = v + bias[n];
                    }
                }
            }
        }
    }
}

// ---------------------------------------------------------------
// Sparse flash attention with block-skip on the nearby mask.
// 128 threads = 4 waves; each wave owns one 16-query tile of one head
// and a private 1KB LDS slice. Fully-masked 32-key blocks are skipped.
// qh, kh: bf16 [head][SEQP][64]; vt: bf16 [head][64][SEQP]
// out: bf16 [SEQP][1024] (heads concatenated along columns)
// ---------------------------------------------------------------
__global__ __launch_bounds__(128)
void k_attn(const bf16_t* __restrict__ qh, const bf16_t* __restrict__ kh,
            const bf16_t* __restrict__ vt, const unsigned char* __restrict__ mask,
            const unsigned char* __restrict__ bm, bf16_t* __restrict__ out) {
    __shared__ bf16_t lp_all[4 * 16 * 32];   // per-wave P-tile slices

    const int wave = threadIdx.x >> 5;
    const int lane = threadIdx.x & 31;
    const int l15  = lane & 15;
    const int hi   = lane >> 4;
    const int qt   = blockIdx.x * 4 + wave;  // 0..167
    const int q0   = qt * 16;
    const int head = blockIdx.y;
    bf16_t* lp = lp_all + wave * (16 * 32);

    const bf16_t* qptr = qh + (size_t)head * SEQP * DH;
    const bf16_t* kptr = kh + (size_t)head * SEQP * DH;
    const bf16_t* vptr = vt + (size_t)head * DH * SEQP;

    // Q fragments for dim-chunks 0..31 and 32..63 (kept in registers)
    v16bf qa[2];
#pragma unroll
    for (int c = 0; c < 2; ++c) {
        const bf16_t* ap = qptr + (size_t)(q0 + l15) * DH + c * 32 + hi * 8;
        qa[c] = ld_frag(ap, ap + 16);
    }

    float run_m[8], run_s[8];
#pragma unroll
    for (int r = 0; r < 8; ++r) { run_m[r] = -3.0e38f; run_s[r] = 0.0f; }
    v8f o[4] = {};

    const float scale = 0.125f;  // 64^-0.5

    for (int kbi = 0; kbi < KB; ++kbi) {
        if (!bm[(size_t)qt * KB + kbi]) continue;   // skip fully-masked block
        const int kb = kbi * 32;

        // ---- S = Q * K^T : load 4 K fragments, then 4 WMMAs ----
        v16bf kf[2][2];
#pragma unroll
        for (int t = 0; t < 2; ++t) {
            int key = kb + t * 16 + l15;
#pragma unroll
            for (int c = 0; c < 2; ++c) {
                const bf16_t* bp = kptr + (size_t)key * DH + c * 32 + hi * 8;
                kf[t][c] = ld_frag(bp, bp + 16);
            }
        }
        v8f s[2];
#pragma unroll
        for (int t = 0; t < 2; ++t) {
            v8f d = {};
            d = wmma_bf16(qa[0], kf[t][0], d);
            d = wmma_bf16(qa[1], kf[t][1], d);
            s[t] = d;
        }

        // ---- scale + nearby mask ----
#pragma unroll
        for (int t = 0; t < 2; ++t) {
            int key = kb + t * 16 + l15;
#pragma unroll
            for (int r = 0; r < 8; ++r) {
                int q = q0 + r + hi * 8;
                bool msk = (q >= SEQ) || (key >= SEQ) ||
                           (mask[(size_t)q * SEQ + key] != 0);
                s[t][r] = msk ? -3.0e38f : s[t][r] * scale;
            }
        }

        // ---- online softmax (rows live across 16 lanes of each half) ----
        float p0[8], p1[8];
#pragma unroll
        for (int r = 0; r < 8; ++r) {
            float mx = fmaxf(s[0][r], s[1][r]);
#pragma unroll
            for (int off = 8; off >= 1; off >>= 1)
                mx = fmaxf(mx, __shfl_xor(mx, off, 32));
            float nm   = fmaxf(run_m[r], mx);
            float corr = __expf(run_m[r] - nm);
            run_m[r]   = nm;
            p0[r] = (s[0][r] < -1.0e37f) ? 0.0f : __expf(s[0][r] - nm);
            p1[r] = (s[1][r] < -1.0e37f) ? 0.0f : __expf(s[1][r] - nm);
            float rs = p0[r] + p1[r];
#pragma unroll
            for (int off = 8; off >= 1; off >>= 1)
                rs += __shfl_xor(rs, off, 32);
            run_s[r] = run_s[r] * corr + rs;
#pragma unroll
            for (int j = 0; j < 4; ++j) o[j][r] *= corr;
        }

        // ---- C-layout P -> A-layout via wave-private LDS slice ----
        __builtin_amdgcn_wave_barrier();
#pragma unroll
        for (int r = 0; r < 8; ++r) {
            lp[(r + hi * 8) * 32 + l15]      = (bf16_t)p0[r];
            lp[(r + hi * 8) * 32 + 16 + l15] = (bf16_t)p1[r];
        }
        __builtin_amdgcn_wave_barrier();
        const bf16_t* pap = &lp[l15 * 32 + hi * 8];
        v16bf pa = ld_frag(pap, pap + 16);

        // ---- O += P(16x32) * V(32x64): load 4 V fragments, then 4 WMMAs ----
        v16bf vf[4];
#pragma unroll
        for (int j = 0; j < 4; ++j) {
            const bf16_t* bp = vptr + (size_t)(j * 16 + l15) * SEQP + kb + hi * 8;
            vf[j] = ld_frag(bp, bp + 16);
        }
#pragma unroll
        for (int j = 0; j < 4; ++j)
            o[j] = wmma_bf16(pa, vf[j], o[j]);
    }

    // ---- epilogue: divide by running sum, store concat-head bf16 ----
    float inv[8];
#pragma unroll
    for (int r = 0; r < 8; ++r) inv[r] = (run_s[r] > 0.0f) ? (1.0f / run_s[r]) : 0.0f;
#pragma unroll
    for (int j = 0; j < 4; ++j) {
#pragma unroll
        for (int r = 0; r < 8; ++r) {
            int m   = q0 + r + hi * 8;
            int col = head * DH + j * 16 + l15;
            out[(size_t)m * DIMK + col] = (bf16_t)(o[j][r] * inv[r]);
        }
    }
}

// ---------------------------------------------------------------
extern "C" void kernel_launch(void* const* d_in, const int* in_sizes, int n_in,
                              void* d_out, int out_size, void* d_ws, size_t ws_size,
                              hipStream_t stream) {
    (void)in_sizes; (void)n_in; (void)out_size; (void)ws_size;

    const float* q  = (const float*)d_in[0];
    const float* k  = (const float*)d_in[1];
    const float* v  = (const float*)d_in[2];
    const float* Wq = (const float*)d_in[3];
    const float* Wk = (const float*)d_in[4];
    const float* Wv = (const float*)d_in[5];
    const float* Wo = (const float*)d_in[6];
    const float* bo = (const float*)d_in[7];
    const unsigned char* mask = (const unsigned char*)d_in[8];

    // ---- workspace carve-out (all offsets 256B aligned) ----
    char* ws = (char*)d_ws;
    size_t off = 0;
    auto carve = [&](size_t bytes) -> char* {
        char* p = ws + off;
        off += (bytes + 255) & ~(size_t)255;
        return p;
    };
    const size_t WB  = (size_t)DIMK * DIMK * sizeof(bf16_t);   // 2 MB
    const size_t ACT = (size_t)SEQP * DIMK * sizeof(bf16_t);   // 5.25 MB

    bf16_t* WqT = (bf16_t*)carve(WB);     // bf16 transposed weights [n][k]
    bf16_t* WkT = (bf16_t*)carve(WB);
    bf16_t* WvT = (bf16_t*)carve(WB);
    bf16_t* WoT = (bf16_t*)carve(WB);
    bf16_t* q_b = (bf16_t*)carve(ACT);
    bf16_t* k_b = (bf16_t*)carve(ACT);
    bf16_t* v_b = (bf16_t*)carve(ACT);
    bf16_t* qh  = (bf16_t*)carve(ACT);    // [head][SEQP][64]
    bf16_t* kh  = (bf16_t*)carve(ACT);    // [head][SEQP][64]
    bf16_t* vt  = (bf16_t*)carve(ACT);    // [head][64][SEQP]
    bf16_t* ao  = (bf16_t*)carve(ACT);    // attention out [SEQP][1024]
    unsigned char* bm = (unsigned char*)carve(QT * KB);  // block mask summary

    // ---- converts (pad activations, transpose weights) + mask summary ----
    {
        int tot_act = SEQP * DIMK;
        int tot_w   = DIMK * DIMK;
        dim3 ba((tot_act + 255) / 256), bw((tot_w + 255) / 256);
        k_pad_bf16<<<ba, 256, 0, stream>>>(q, q_b, SEQ, SEQP, DIMK);
        k_pad_bf16<<<ba, 256, 0, stream>>>(k, k_b, SEQ, SEQP, DIMK);
        k_pad_bf16<<<ba, 256, 0, stream>>>(v, v_b, SEQ, SEQP, DIMK);
        k_wtrans_bf16<<<bw, 256, 0, stream>>>(Wq, WqT);
        k_wtrans_bf16<<<bw, 256, 0, stream>>>(Wk, WkT);
        k_wtrans_bf16<<<bw, 256, 0, stream>>>(Wv, WvT);
        k_wtrans_bf16<<<bw, 256, 0, stream>>>(Wo, WoT);
        k_maskblk<<<(QT * KB + 255) / 256, 256, 0, stream>>>(mask, bm);
    }

    // ---- projections (LDS-free WMMA GEMM) ----
    dim3 gg(SEQP / 128, DIMK / 64);   // (21, 16)
    k_gemm<<<gg, 128, 0, stream>>>(q_b, WqT, qh, nullptr, 0);
    k_gemm<<<gg, 128, 0, stream>>>(k_b, WkT, kh, nullptr, 0);
    k_gemm<<<gg, 128, 0, stream>>>(v_b, WvT, vt, nullptr, 1);

    // ---- attention (block-sparse flash) ----
    dim3 ga(QT / 4, HEADS);           // (42, 16)
    k_attn<<<ga, 128, 0, stream>>>(qh, kh, vt, mask, bm, ao);

    // ---- output projection + bias (f32 to d_out) ----
    k_gemm<<<gg, 128, 0, stream>>>(ao, WoT, d_out, bo, 2);
}
